// HgnnEncoder_36017595744531
// MI455X (gfx1250) — matrix operand order repaired and checked by reading, not compiled, not run on hardware
//
#include <hip/hip_runtime.h>
#include <hip/hip_bf16.h>
#include <math.h>

#define N_NODES 20000
#define N_EDGES 5000
#define NNZ     640000
#define IN_DIM  256
#define OUT_DIM 256
#define HID     768
#define K_CODES 1024
#define N_LAYERS 3
#define BN_EPS  1e-5f
#define COMMIT  0.5f

typedef __bf16 bf16_t;
typedef __attribute__((ext_vector_type(16))) __bf16 v16bf;
typedef __attribute__((ext_vector_type(8)))  float  v8f;

#define CDIV(a,b) (((a)+(b)-1)/(b))

// ---------------------------------------------------------------- utilities
__global__ __launch_bounds__(256) void k_zero(float* p, long n) {
  long i = (long)blockIdx.x * 256 + threadIdx.x;
  if (i < n) p[i] = 0.f;
}
__global__ __launch_bounds__(256) void k_copy(float* dst, const float* src, long n) {
  long i = (long)blockIdx.x * 256 + threadIdx.x;
  if (i < n) dst[i] = src[i];
}
__global__ __launch_bounds__(256) void k_add(float* dst, const float* a, const float* b, long n) {
  long i = (long)blockIdx.x * 256 + threadIdx.x;
  if (i < n) dst[i] = a[i] + b[i];
}

// Pre-swizzle weights into the layout this kernel's WMMA B loads use (contiguous k order):
//   Bp[(k>>5)*N*32 + n*32 + (k&31)] = W[k][n]
// so each lane's 16x16x32 B fragment (K = e + 16*hf) is 32 contiguous bytes.
__global__ __launch_bounds__(256)
void k_cast_prep(bf16_t* __restrict__ dst, const float* __restrict__ src, int K, int N) {
  long i = (long)blockIdx.x * 256 + threadIdx.x;
  if (i < (long)K * N) {
    int k = (int)(i / N), n = (int)(i % N);
    dst[(((long)(k >> 5) * N + n) << 5) + (k & 31)] = (bf16_t)src[i];
  }
}
// emb [K_CODES, HID] row-major; B = emb^T (K=HID, N=K_CODES), same prepped layout.
__global__ __launch_bounds__(256)
void k_transpose_prep(bf16_t* __restrict__ dst, const float* __restrict__ src) {
  long i = (long)blockIdx.x * 256 + threadIdx.x;
  if (i < (long)K_CODES * HID) {
    int n = (int)(i / HID), k = (int)(i % HID);
    dst[(((long)(k >> 5) * K_CODES + n) << 5) + (k & 31)] = (bf16_t)src[i];
  }
}

__device__ inline float block_reduce_sum(float v, float* sm) {
  int t = threadIdx.x;
  sm[t] = v; __syncthreads();
  for (int s = 128; s > 0; s >>= 1) { if (t < s) sm[t] += sm[t + s]; __syncthreads(); }
  float r = sm[0]; __syncthreads();
  return r;
}

// ---------------------------------------------------------------- WMMA GEMM
// C[M,Nn] = act( (A*bnScale+bnShift)[M,K] @ B[K,Nn] + bias )
// A fp32 row-major; Bp bf16 in prepped layout; C fp32. K%32==0, Nn%64==0.
#define BM 128
#define BN 64
#define BK 32
#define ASTR 40   // bf16 elems per LDS row (80B, 16B-aligned)

__global__ __launch_bounds__(256)
void k_wmma_gemm(const float* __restrict__ A, const bf16_t* __restrict__ Bp,
                 float* __restrict__ C, int M, int Nn, int K,
                 const float* __restrict__ bnScale, const float* __restrict__ bnShift,
                 const float* __restrict__ bias, int relu) {
  __shared__ bf16_t As[2][BM * ASTR];
  const int t    = threadIdx.x;
  const int bm   = blockIdx.x * BM;
  const int bn   = blockIdx.y * BN;
  const int lane = t & 31;
  const int wave = t >> 5;
  const int wm   = (wave >> 1) * 32;   // 4 wave-rows
  const int wn   = (wave & 1) * 32;    // 2 wave-cols
  const int hf   = lane >> 4;          // K-half selector
  const int rr   = lane & 15;
  const bool hasBN = (bnScale != nullptr);

  const int arow = t >> 3;             // 0..31, +32*j
  const int acol = (t & 7) << 2;       // 0..28 step 4

  v8f c00 = {}, c01 = {}, c10 = {}, c11 = {};
  const int nk = K >> 5;

  float4 vreg[4];

  // fetch A tile kb into registers (branch-free edge handling, fused BN affine)
  auto fetch = [&](int kb) {
    const int k0 = kb << 5;
#pragma unroll
    for (int j = 0; j < 4; ++j) {
      int grow = bm + arow + 32 * j;
      int crow = grow < M ? grow : M - 1;
      float4 v = *reinterpret_cast<const float4*>(&A[(long)crow * K + k0 + acol]);
      if (grow >= M) v = make_float4(0.f, 0.f, 0.f, 0.f);
      if (hasBN) {
        v.x = v.x * bnScale[k0 + acol + 0] + bnShift[k0 + acol + 0];
        v.y = v.y * bnScale[k0 + acol + 1] + bnShift[k0 + acol + 1];
        v.z = v.z * bnScale[k0 + acol + 2] + bnShift[k0 + acol + 2];
        v.w = v.w * bnScale[k0 + acol + 3] + bnShift[k0 + acol + 3];
      }
      vreg[j] = v;
    }
  };
  // commit registers -> LDS buffer (packed bf16, 8B stores)
  auto commit = [&](int buf) {
#pragma unroll
    for (int j = 0; j < 4; ++j) {
      union { bf16_t h[4]; uint2 u; } p;
      p.h[0] = (bf16_t)vreg[j].x; p.h[1] = (bf16_t)vreg[j].y;
      p.h[2] = (bf16_t)vreg[j].z; p.h[3] = (bf16_t)vreg[j].w;
      *reinterpret_cast<uint2*>(&As[buf][(arow + 32 * j) * ASTR + acol]) = p.u;
    }
  };

  fetch(0);
  commit(0);
  __syncthreads();

  for (int kb = 0; kb < nk; ++kb) {
    const int cur = kb & 1;
    if (kb + 1 < nk) {
      fetch(kb + 1);
      __builtin_prefetch(&Bp[(((long)(kb + 1) * Nn + bn + wn + rr) << 5)], 0, 3);
    }

    // A fragments: two contiguous 16B runs per lane (k = 16hf+e , 16hf+8+e)
    union U { uint4 q[2]; v16bf v; };
    U ua0, ua1;
    const bf16_t* arow0 = &As[cur][(wm + rr) * ASTR];
    const bf16_t* arow1 = &As[cur][(wm + 16 + rr) * ASTR];
    ua0.q[0] = *reinterpret_cast<const uint4*>(arow0 + (hf << 4));
    ua0.q[1] = *reinterpret_cast<const uint4*>(arow0 + 8 + (hf << 4));
    ua1.q[0] = *reinterpret_cast<const uint4*>(arow1 + (hf << 4));
    ua1.q[1] = *reinterpret_cast<const uint4*>(arow1 + 8 + (hf << 4));

    // B fragments: single 32B contiguous vector load from prepped global layout
    const bf16_t* Bblk = Bp + (((long)kb * Nn) << 5);
    v16bf b0 = *reinterpret_cast<const v16bf*>(&Bblk[(((long)(bn + wn + rr)) << 5) + (hf << 4)]);
    v16bf b1 = *reinterpret_cast<const v16bf*>(&Bblk[(((long)(bn + wn + 16 + rr)) << 5) + (hf << 4)]);

    c00 = __builtin_amdgcn_wmma_f32_16x16x32_bf16(false, ua0.v, false, b0, (short)0, c00, false, false);
    c01 = __builtin_amdgcn_wmma_f32_16x16x32_bf16(false, ua0.v, false, b1, (short)0, c01, false, false);
    c10 = __builtin_amdgcn_wmma_f32_16x16x32_bf16(false, ua1.v, false, b0, (short)0, c10, false, false);
    c11 = __builtin_amdgcn_wmma_f32_16x16x32_bf16(false, ua1.v, false, b1, (short)0, c11, false, false);

    if (kb + 1 < nk) commit(cur ^ 1);
    __syncthreads();
  }

  // store per C/D layout: lanes 0-15 -> M=r, lanes 16-31 -> M=8+r; N = lane&15
  const int cn = lane & 15;
  const int rbase = (lane >> 4) << 3;
  v8f accs[4] = {c00, c01, c10, c11};
#pragma unroll
  for (int ij = 0; ij < 4; ++ij) {
    int i = ij >> 1, j = ij & 1;
    int col = bn + wn + j * 16 + cn;
    float bv = bias ? bias[col] : 0.f;
#pragma unroll
    for (int r = 0; r < 8; ++r) {
      int row = bm + wm + i * 16 + rbase + r;
      if (row < M) {
        float v = accs[ij][r] + bv;
        if (relu) v = fmaxf(v, 0.f);
        C[(long)row * Nn + col] = v;
      }
    }
  }
}

// ---------------------------------------------------------------- BatchNorm
__global__ __launch_bounds__(256)
void k_bn_part(const float* __restrict__ A, int M, int C, float* s1, float* s2) {
  int rb = blockIdx.x * 128;
  int rend = min(rb + 128, M);
  for (int c = threadIdx.x; c < C; c += 256) {
    float a = 0.f, b = 0.f;
    for (int r = rb; r < rend; ++r) { float x = A[(long)r * C + c]; a += x; b += x * x; }
    atomicAdd(&s1[c], a); atomicAdd(&s2[c], b);
  }
}
__global__ __launch_bounds__(256)
void k_bn_finalize(const float* s1, const float* s2, const float* g, const float* b,
                   float* scale, float* shift, int C, float M) {
  int c = blockIdx.x * 256 + threadIdx.x;
  if (c < C) {
    float m = s1[c] / M;
    float var = s2[c] / M - m * m;
    float sc = g[c] * rsqrtf(var + BN_EPS);
    scale[c] = sc;
    shift[c] = b[c] - m * sc;
  }
}

// ---------------------------------------------------------------- graph ops
__global__ __launch_bounds__(256)
void k_degrees(const int* ni, const int* ei, float* dn, float* de) {
  int i = blockIdx.x * 256 + threadIdx.x;
  if (i < NNZ) { atomicAdd(&dn[ni[i]], 1.f); atomicAdd(&de[ei[i]], 1.f); }
}
__global__ __launch_bounds__(256)
void k_recip(float* d, int n) {
  int i = blockIdx.x * 256 + threadIdx.x;
  if (i < n) d[i] = d[i] > 0.f ? 1.f / d[i] : 0.f;
}
// Me[e] += Xs[n]  over incidences
__global__ __launch_bounds__(256)
void k_scatter_edge(const float* __restrict__ Xs, const int* __restrict__ ni,
                    const int* __restrict__ ei, float* __restrict__ Me, int C) {
  long i = (long)blockIdx.x * 256 + threadIdx.x;
  int c4 = C >> 2;
  if (i >= (long)NNZ * c4) return;
  int pair = (int)(i / c4), c = (int)(i % c4) << 2;
  int n = ni[pair], e = ei[pair];
  float4 v = *reinterpret_cast<const float4*>(&Xs[(long)n * C + c]);
  float* dst = &Me[(long)e * C + c];
  atomicAdd(dst + 0, v.x); atomicAdd(dst + 1, v.y);
  atomicAdd(dst + 2, v.z); atomicAdd(dst + 3, v.w);
}
// Acc[n] += Me[e] * Binv[e]
__global__ __launch_bounds__(256)
void k_gather_node(const float* __restrict__ Me, const float* __restrict__ Binv,
                   const int* __restrict__ ni, const int* __restrict__ ei,
                   float* __restrict__ Acc, int C) {
  long i = (long)blockIdx.x * 256 + threadIdx.x;
  int c4 = C >> 2;
  if (i >= (long)NNZ * c4) return;
  int pair = (int)(i / c4), c = (int)(i % c4) << 2;
  int n = ni[pair], e = ei[pair];
  float s = Binv[e];
  float4 v = *reinterpret_cast<const float4*>(&Me[(long)e * C + c]);
  float* dst = &Acc[(long)n * C + c];
  atomicAdd(dst + 0, v.x * s); atomicAdd(dst + 1, v.y * s);
  atomicAdd(dst + 2, v.z * s); atomicAdd(dst + 3, v.w * s);
}
// fused: v = relu(Acc*Dinv + bias); either Acc = v*gate (layer path) or Out += v (final path)
__global__ __launch_bounds__(256)
void k_finish_msg(float* __restrict__ Acc, const float* __restrict__ Dinv,
                  const float* __restrict__ bias, const float* __restrict__ gate,
                  float* __restrict__ outAdd, int N, int C) {
  long i = (long)blockIdx.x * 256 + threadIdx.x;
  if (i >= (long)N * C) return;
  int n = (int)(i / C), c = (int)(i % C);
  float v = fmaxf(Acc[i] * Dinv[n] + bias[c], 0.f);
  if (outAdd) outAdd[i] += v;
  else        Acc[i] = v * gate[n];
}

// ---------------------------------------------------------------- gate
__global__ __launch_bounds__(256)
void k_gate(const float* __restrict__ X, const float* __restrict__ sc,
            const float* __restrict__ sh, const float* __restrict__ gw,
            const float* __restrict__ gb, float* __restrict__ gate) {
  __shared__ float sm[256];
  int n = blockIdx.x;
  float acc = 0.f;
  for (int c = threadIdx.x; c < HID; c += 256)
    acc += (X[(long)n * HID + c] * sc[c] + sh[c]) * gw[c];
  acc = block_reduce_sum(acc, sm);
  if (threadIdx.x == 0) gate[n] = 1.f / (1.f + __expf(-(acc + gb[0])));
}

// ---------------------------------------------------------------- VQ
__device__ inline float hash_u01(unsigned x) {
  x ^= x >> 17; x *= 0xed5ad4bbu; x ^= x >> 11; x *= 0xac4c1b51u;
  x ^= x >> 15; x *= 0x31848babu; x ^= x >> 14;
  return (float)(x >> 8) * (1.0f / 16777216.0f);
}
__global__ __launch_bounds__(256)
void k_enorm(const float* __restrict__ emb, float* __restrict__ en) {
  __shared__ float sm[256];
  int k = blockIdx.x;
  float s = 0.f;
  for (int c = threadIdx.x; c < HID; c += 256) { float e = emb[(long)k * HID + c]; s += e * e; }
  s = block_reduce_sum(s, sm);
  if (threadIdx.x == 0) en[k] = s;
}
// logits = 2*Sc - ||x||^2 - ||e||^2 ; argmax(logits+gumbel) ; kld accumulation
__global__ __launch_bounds__(256)
void k_vq_select(const float* __restrict__ Sc, const float* __restrict__ inp,
                 const float* __restrict__ en, int layer, int* __restrict__ codes,
                 float* __restrict__ hist, float* __restrict__ accs) {
  __shared__ float sv[256];
  __shared__ int   si[256];
  int n = blockIdx.x, t = threadIdx.x;
  float rn = 0.f;
  for (int c = t; c < HID; c += 256) { float x = inp[(long)n * HID + c]; rn += x * x; }
  rn = block_reduce_sum(rn, sv);
  float best = -3.4e38f, kl = 0.f;
  int bi = 0;
  for (int k = t; k < K_CODES; k += 256) {
    float logit = 2.f * Sc[(long)n * K_CODES + k] - rn - en[k];
    float p = 1.f / (1.f + __expf(-logit));
    kl += p * __logf(fmaxf(p, 1e-8f));
    unsigned seed = (unsigned)(n * K_CODES + k) + (unsigned)layer * 0x9e3779b9u;
    float u = hash_u01(seed);
    float g = -__logf(-__logf(u + 1e-20f) + 1e-20f);
    float s = logit + g;
    if (s > best) { best = s; bi = k; }
  }
  kl = block_reduce_sum(kl, sv);
  sv[t] = best; si[t] = bi; __syncthreads();
  for (int s = 128; s > 0; s >>= 1) {
    if (t < s && sv[t + s] > sv[t]) { sv[t] = sv[t + s]; si[t] = si[t + s]; }
    __syncthreads();
  }
  if (t == 0) {
    codes[n] = si[0];
    atomicAdd(&hist[si[0]], 1.f);
    atomicAdd(&accs[0], kl);
  }
}
// q = emb[code]; Xb += q; accumulate e_latent
__global__ __launch_bounds__(256)
void k_vq_apply(float* __restrict__ Xb, const float* __restrict__ inp,
                const float* __restrict__ emb, const int* __restrict__ codes,
                float* __restrict__ acc_el) {
  __shared__ float sm[256];
  int n = blockIdx.x, k = codes[n];
  float d2 = 0.f;
  for (int c = threadIdx.x; c < HID; c += 256) {
    float q = emb[(long)k * HID + c];
    float x = inp[(long)n * HID + c];
    Xb[(long)n * HID + c] += q;
    float d = q - x; d2 += d * d;
  }
  d2 = block_reduce_sum(d2, sm);
  if (threadIdx.x == 0) atomicAdd(acc_el, d2);
}
__global__ __launch_bounds__(256)
void k_vq_finalize(const float* accs, const float* hist, float* out2) {
  __shared__ float sm[256];
  float s = 0.f;
  for (int k = threadIdx.x; k < K_CODES; k += 256) {
    float a = hist[k] / (float)N_NODES;
    s += a * __logf(a + 1e-10f);
  }
  s = block_reduce_sum(s, sm);
  if (threadIdx.x == 0) {
    float kld = accs[0] / (float)N_NODES;
    float el  = accs[1] / ((float)N_NODES * (float)HID);
    out2[0] = COMMIT * (kld + el * (kld / fmaxf(el, 1e-8f)));
    out2[1] = __expf(-s);
  }
}

// ---------------------------------------------------------------- launcher
extern "C" void kernel_launch(void* const* d_in, const int* in_sizes, int n_in,
                              void* d_out, int out_size, void* d_ws, size_t ws_size,
                              hipStream_t stream) {
  const float* X     = (const float*)d_in[0];
  const int*   nidx  = (const int*)d_in[1];
  const int*   eidx  = nidx + NNZ;
  const float* W_up  = (const float*)d_in[2];
  const float* b_up  = (const float*)d_in[3];
  const float* bn1_g = (const float*)d_in[4];
  const float* bn1_b = (const float*)d_in[5];
  const float* hgW   = (const float*)d_in[6];
  const float* hgb   = (const float*)d_in[7];
  const float* bng_g = (const float*)d_in[8];
  const float* bng_b = (const float*)d_in[9];
  const float* gateW = (const float*)d_in[10];
  const float* gateb = (const float*)d_in[11];
  const float* emb   = (const float*)d_in[12];
  const float* bnd_g = (const float*)d_in[13];
  const float* bnd_b = (const float*)d_in[14];
  const float* Wd    = (const float*)d_in[15];
  const float* bd    = (const float*)d_in[16];
  const float* bnc_g = (const float*)d_in[17];
  const float* bnc_b = (const float*)d_in[18];
  const float* hgWd  = (const float*)d_in[19];
  const float* hgbd  = (const float*)d_in[20];
  float* out = (float*)d_out;

  char* base = (char*)d_ws; size_t off = 0;
  auto alloc = [&](size_t bytes) -> void* {
    off = (off + 255) & ~(size_t)255;
    void* p = base + off; off += bytes; return p;
  };
  float*  Xb   = (float*)alloc((size_t)N_NODES * HID * 4);
  float*  Id   = (float*)alloc((size_t)N_NODES * HID * 4);
  float*  T1   = (float*)alloc((size_t)N_NODES * HID * 4);
  float*  T2   = (float*)alloc((size_t)N_NODES * HID * 4);
  float*  Me   = (float*)alloc((size_t)N_EDGES * HID * 4);
  float*  Sc   = (float*)alloc((size_t)N_NODES * K_CODES * 4);
  bf16_t* Bw   = (bf16_t*)alloc((size_t)HID * K_CODES * 2);
  float*  dn   = (float*)alloc((size_t)N_NODES * 4);
  float*  de   = (float*)alloc((size_t)N_EDGES * 4);
  float*  gatev= (float*)alloc((size_t)N_NODES * 4);
  int*    codes= (int*)  alloc((size_t)N_NODES * 4);
  float*  hist = (float*)alloc((size_t)K_CODES * 4);
  float*  en   = (float*)alloc((size_t)K_CODES * 4);
  float*  s1   = (float*)alloc((size_t)2 * HID * 4); float* s2 = s1 + HID;
  float*  scA  = (float*)alloc((size_t)HID * 4);
  float*  shA  = (float*)alloc((size_t)HID * 4);
  float*  accs = (float*)alloc(4 * 4);

  auto gemm = [&](const float* A, const bf16_t* B, float* C, int M, int Nn, int K,
                  const float* sc, const float* sh, const float* bias, int relu) {
    dim3 g(CDIV(M, BM), Nn / BN);
    k_wmma_gemm<<<g, 256, 0, stream>>>(A, B, C, M, Nn, K, sc, sh, bias, relu);
  };
  auto prep = [&](const float* W, int K, int N) {
    k_cast_prep<<<CDIV((long)K * N, 256), 256, 0, stream>>>(Bw, W, K, N);
  };
  auto zero = [&](float* p, long n) { k_zero<<<CDIV(n, 256), 256, 0, stream>>>(p, n); };
  auto bn = [&](const float* A, const float* g, const float* b) {
    zero(s1, 2 * HID);
    k_bn_part<<<CDIV(N_NODES, 128), 256, 0, stream>>>(A, N_NODES, HID, s1, s2);
    k_bn_finalize<<<CDIV(HID, 256), 256, 0, stream>>>(s1, s2, g, b, scA, shA, HID, (float)N_NODES);
  };
  auto hconv_mp = [&](float* XW, int C, const float* biasv, float* outAdd) {
    long sc_total = (long)NNZ * (C >> 2);
    zero(Me, (long)N_EDGES * C);
    k_scatter_edge<<<CDIV(sc_total, 256), 256, 0, stream>>>(XW, nidx, eidx, Me, C);
    zero(T2, (long)N_NODES * C);
    k_gather_node<<<CDIV(sc_total, 256), 256, 0, stream>>>(Me, de, nidx, eidx, T2, C);
    k_finish_msg<<<CDIV((long)N_NODES * C, 256), 256, 0, stream>>>(
        T2, dn, biasv, gatev, outAdd, N_NODES, C);
  };

  // degrees -> Dinv/Binv (stored inverted in dn/de)
  zero(dn, N_NODES); zero(de, N_EDGES);
  k_degrees<<<CDIV(NNZ, 256), 256, 0, stream>>>(nidx, eidx, dn, de);
  k_recip<<<CDIV(N_NODES, 256), 256, 0, stream>>>(dn, N_NODES);
  k_recip<<<CDIV(N_EDGES, 256), 256, 0, stream>>>(de, N_EDGES);

  // Xb = relu(X @ W_up + b_up); identity = Xb
  prep(W_up, IN_DIM, HID);
  gemm(X, Bw, Xb, N_NODES, HID, IN_DIM, nullptr, nullptr, b_up, 1);
  k_copy<<<CDIV((long)N_NODES * HID, 256), 256, 0, stream>>>(Id, Xb, (long)N_NODES * HID);

  for (int i = 0; i < N_LAYERS; ++i) {
    // T1 = bn1(Xb) @ hgW[i]  (BN fused into WMMA A-load)
    bn(Xb, bn1_g + i * HID, bn1_b + i * HID);
    prep(hgW + (long)i * HID * HID, HID, HID);
    gemm(Xb, Bw, T1, N_NODES, HID, HID, scA, shA, nullptr, 0);
    // gate = sigmoid(bng(Xb) @ gateW + gateb)
    bn(Xb, bng_g + i * HID, bng_b + i * HID);
    k_gate<<<N_NODES, 256, 0, stream>>>(Xb, scA, shA, gateW + (long)i * HID, gateb + i, gatev);
    // message passing; T2 = relu(hconv)*gate  (bias+relu+gate fused)
    hconv_mp(T1, HID, hgb + i * HID, nullptr);
    // VQ: scores via WMMA GEMM against emb^T, then straight-through gather
    const float* emb_i = emb + (long)i * K_CODES * HID;
    k_transpose_prep<<<CDIV((long)K_CODES * HID, 256), 256, 0, stream>>>(Bw, emb_i);
    gemm(T2, Bw, Sc, N_NODES, K_CODES, HID, nullptr, nullptr, nullptr, 0);
    k_enorm<<<K_CODES, 256, 0, stream>>>(emb_i, en);
    zero(hist, K_CODES); zero(accs, 4);
    k_vq_select<<<N_NODES, 256, 0, stream>>>(Sc, T2, en, i, codes, hist, accs);
    k_vq_apply<<<N_NODES, 256, 0, stream>>>(Xb, T2, emb_i, codes, accs + 1);
  }

  // Xo = bnd(Xb + identity) @ Wd + bd  -> out
  k_add<<<CDIV((long)N_NODES * HID, 256), 256, 0, stream>>>(T1, Xb, Id, (long)N_NODES * HID);
  bn(T1, bnd_g, bnd_b);
  prep(Wd, HID, OUT_DIM);
  gemm(T1, Bw, out, N_NODES, OUT_DIM, HID, scA, shA, bd, 0);

  // Xc = relu(hconv(bnc(Xb), hgWd, hgbd)) -> out +=
  bn(Xb, bnc_g, bnc_b);
  prep(hgWd, HID, OUT_DIM);
  gemm(Xb, Bw, T1, N_NODES, OUT_DIM, HID, scA, shA, nullptr, 0);
  hconv_mp(T1, OUT_DIM, hgbd, out);

  // loss & perplexity from the last layer's accumulators
  k_vq_finalize<<<1, 256, 0, stream>>>(accs, hist, out + (long)N_NODES * OUT_DIM);
}
